// EMAMixing_54107997995601
// MI455X (gfx1250) — matrix operand order repaired and checked by reading, not compiled, not run on hardware
//
#include <hip/hip_runtime.h>
#include <stdint.h>

#define L_SEQ 8192
#define DIM   2048
#define CHUNK 128
#define NCHUNK (L_SEQ / CHUNK)   // 64

// GEMM tiling
#define KSTEP   64
#define NSTAGE  (DIM / KSTEP)    // 32
#define LSTRIDE 72               // padded LDS row stride (elements, 144B: 16B-aligned, bank-friendly)
#define TILE_ELEMS (128 * LSTRIDE)

typedef __attribute__((ext_vector_type(16))) __bf16 v16bf;
typedef __attribute__((ext_vector_type(8)))  float  v8f;
typedef int v4i_ __attribute__((vector_size(16)));   // matches builtin param pointee

#define AS1 __attribute__((address_space(1)))
#define AS3 __attribute__((address_space(3)))

static __device__ __forceinline__ uint16_t f32_to_bf16(float f) {
    uint32_t u = __float_as_uint(f);
    u += 0x7fffu + ((u >> 16) & 1u);      // round-to-nearest-even
    return (uint16_t)(u >> 16);
}

static __device__ __forceinline__ float sigmoidf_(float p) {
    return 1.0f / (1.0f + __expf(-p));
}

// ---- gfx1250 async global->LDS copy (ASYNCcnt-tracked), with sync fallback ----
static __device__ __forceinline__ void async_copy16(const uint16_t* g, uint16_t* l) {
#if __has_builtin(__builtin_amdgcn_global_load_async_to_lds_b128)
    __builtin_amdgcn_global_load_async_to_lds_b128(
        (AS1 v4i_*)(uintptr_t)(const void*)g,
        (AS3 v4i_*)l,
        0, 0);
#else
    *(uint4*)l = *(const uint4*)g;
#endif
}

static __device__ __forceinline__ void wait_async_le8() {
#if __has_builtin(__builtin_amdgcn_s_wait_asynccnt)
    __builtin_amdgcn_s_wait_asynccnt(8);
#else
    asm volatile("s_wait_asynccnt 0x8" ::: "memory");
#endif
}
static __device__ __forceinline__ void wait_async_le0() {
#if __has_builtin(__builtin_amdgcn_s_wait_asynccnt)
    __builtin_amdgcn_s_wait_asynccnt(0);
#else
    asm volatile("s_wait_asynccnt 0x0" ::: "memory");
#endif
}

// ---------------- Pass 1: chunk-local EMA carries (zero initial state) ------
__global__ void ema_carry_kernel(const float* __restrict__ x,
                                 const float* __restrict__ factor_param,
                                 float* __restrict__ carry) {
    const int d = blockIdx.x * blockDim.x + threadIdx.x;   // channel
    const int c = blockIdx.y;                              // chunk
    const float f   = sigmoidf_(factor_param[d]);
    const float omf = 1.0f - f;
    const float* xp = x + (size_t)c * CHUNK * DIM + d;
    float s = 0.0f;
#pragma unroll 8
    for (int t = 0; t < CHUNK; ++t)
        s = fmaf(f, s, omf * xp[(size_t)t * DIM]);
    carry[c * DIM + d] = s;
}

// ---------------- Pass 2: per-channel prefix over chunk carries -------------
__global__ void ema_prefix_kernel(const float* __restrict__ x_mix_last,
                                  const float* __restrict__ factor_param,
                                  const float* __restrict__ carry,
                                  float* __restrict__ in_state) {
    const int d = blockIdx.x * blockDim.x + threadIdx.x;
    const float f = sigmoidf_(factor_param[d]);
    float fC = f;                      // f^128 via 7 squarings
#pragma unroll
    for (int i = 0; i < 7; ++i) fC *= fC;
    float state = x_mix_last[d];
    for (int c = 0; c < NCHUNK; ++c) {
        in_state[c * DIM + d] = state;
        state = fmaf(fC, state, carry[c * DIM + d]);
    }
}

// ---------------- Pass 3: apply scan with correct state, emit bf16 ----------
__global__ void ema_apply_kernel(const float* __restrict__ x,
                                 const float* __restrict__ factor_param,
                                 const float* __restrict__ in_state,
                                 uint16_t* __restrict__ xmix_bf16) {
    const int d = blockIdx.x * blockDim.x + threadIdx.x;
    const int c = blockIdx.y;
    const float f   = sigmoidf_(factor_param[d]);
    const float omf = 1.0f - f;
    float m = in_state[c * DIM + d];
    const float* xp = x + (size_t)c * CHUNK * DIM + d;
    uint16_t* op = xmix_bf16 + (size_t)c * CHUNK * DIM + d;
#pragma unroll 8
    for (int t = 0; t < CHUNK; ++t) {
        m = fmaf(f, m, omf * xp[(size_t)t * DIM]);
        op[(size_t)t * DIM] = f32_to_bf16(m);
    }
}

// ---------------- W -> bf16 (row-major [D,D]) -------------------------------
__global__ void bf16_convert4_kernel(const float4* __restrict__ src,
                                     ushort4* __restrict__ dst) {
    const int i = blockIdx.x * blockDim.x + threadIdx.x;
    float4 v = src[i];
    ushort4 o;
    o.x = f32_to_bf16(v.x); o.y = f32_to_bf16(v.y);
    o.z = f32_to_bf16(v.z); o.w = f32_to_bf16(v.w);
    dst[i] = o;
}

// Stage one 128x64 A tile and 64x128 B tile (stored as 128 cols x 64 k) into LDS.
// 8 async 16B copies per thread (256 threads): A = 128 rows x 8 segs, B likewise.
static __device__ __forceinline__ void stage_tile(const uint16_t* __restrict__ A,
                                                  const uint16_t* __restrict__ B,
                                                  int m_blk, int n_blk, int k0,
                                                  uint16_t* sA, uint16_t* sB, int tid) {
#pragma unroll
    for (int r = 0; r < 4; ++r) {
        int s = tid + r * 256;          // 0..1023
        int row = s >> 3, seg = s & 7;  // seg = 8-element (16B) chunk along K
        async_copy16(A + (size_t)(m_blk + row) * DIM + k0 + seg * 8,
                     sA + row * LSTRIDE + seg * 8);
    }
#pragma unroll
    for (int r = 0; r < 4; ++r) {
        int s = tid + r * 256;
        int col = s >> 3, seg = s & 7;
        async_copy16(B + (size_t)(n_blk + col) * DIM + k0 + seg * 8,
                     sB + col * LSTRIDE + seg * 8);
    }
}

// ---------------- GEMM: out[L,D] f32 = xmix_bf16 @ W_bf16^T -----------------
// Block tile 128M x 128N, 8 waves (2x4), each wave 64M x 32N.
// Double-buffered LDS staging via async global->LDS, K-stage 64 (2 WMMA half-steps).
__global__ void __launch_bounds__(256)
gemm_bf16_wmma_kernel(const uint16_t* __restrict__ A,   // xmix bf16 [L, D]
                      const uint16_t* __restrict__ B,   // W    bf16 [D, D]
                      float* __restrict__ C) {          // out  f32  [L, D]
    extern __shared__ uint16_t smem[];                  // [2 A bufs][2 B bufs]
    uint16_t* sA0 = smem;                               // A: buf*TILE_ELEMS
    uint16_t* sB0 = smem + 2 * TILE_ELEMS;              // B: buf*TILE_ELEMS

    const int tid    = threadIdx.x;
    const int lane   = tid & 31;
    const int wave   = tid >> 5;
    const int lane16 = lane & 15;
    const int laneHi = lane >> 4;

    const int m_blk = blockIdx.y * 128;
    const int n_blk = blockIdx.x * 128;
    const int wm = (wave >> 2) * 64;    // wave M offset within block tile
    const int wn = (wave & 3) * 32;     // wave N offset within block tile

    v8f acc[4][2];
#pragma unroll
    for (int i = 0; i < 4; ++i)
#pragma unroll
        for (int j = 0; j < 2; ++j)
#pragma unroll
            for (int e = 0; e < 8; ++e) acc[i][j][e] = 0.0f;

    // Preload stage 0 into buffer 0.
    stage_tile(A, B, m_blk, n_blk, 0, sA0, sB0, tid);

    for (int kk = 0; kk < NSTAGE; ++kk) {
        const int buf = kk & 1;
        if (kk + 1 < NSTAGE) {
            stage_tile(A, B, m_blk, n_blk, (kk + 1) * KSTEP,
                       sA0 + (buf ^ 1) * TILE_ELEMS,
                       sB0 + (buf ^ 1) * TILE_ELEMS, tid);
            wait_async_le8();           // current stage landed; next 8 stay in flight
        } else {
            wait_async_le0();
        }
        __syncthreads();

        const uint16_t* tA = sA0 + buf * TILE_ELEMS;
        const uint16_t* tB = sB0 + buf * TILE_ELEMS;
#pragma unroll
        for (int h = 0; h < 2; ++h) {   // two 32-deep WMMA half-steps per stage
            const int kh = h * 32;
            v16bf af[4], bf[2];
#pragma unroll
            for (int i = 0; i < 4; ++i) {
                // A frag (16x32 bf16): lane m holds K=hi*8..+7 (regs 0-3), K=16+hi*8..+7 (regs 4-7)
                const uint16_t* p = tA + (wm + i * 16 + lane16) * LSTRIDE + kh + laneHi * 8;
                union { uint4 u[2]; v16bf v; } cv;
                cv.u[0] = *(const uint4*)(p);
                cv.u[1] = *(const uint4*)(p + 16);
                af[i] = cv.v;
            }
#pragma unroll
            for (int j = 0; j < 2; ++j) {
                // B frag (32x16): column lane16 holds K=hi*8..+7 (regs 0-3), K=16+hi*8..+7 (regs 4-7)
                const uint16_t* p = tB + (wn + j * 16 + lane16) * LSTRIDE + kh + laneHi * 8;
                union { uint4 u[2]; v16bf v; } cv;
                cv.u[0] = *(const uint4*)(p);
                cv.u[1] = *(const uint4*)(p + 16);
                bf[j] = cv.v;
            }
#pragma unroll
            for (int i = 0; i < 4; ++i)
#pragma unroll
                for (int j = 0; j < 2; ++j)
                    acc[i][j] = __builtin_amdgcn_wmma_f32_16x16x32_bf16(
                        false, af[i], false, bf[j],
                        (short)0, acc[i][j], false, false);
        }
        __syncthreads();
    }

    // Epilogue: C/D layout — VGPR r: M = r + laneHi*8, N = lane16.
#pragma unroll
    for (int i = 0; i < 4; ++i)
#pragma unroll
        for (int j = 0; j < 2; ++j) {
            const int n = n_blk + wn + j * 16 + lane16;
#pragma unroll
            for (int r = 0; r < 8; ++r) {
                const int m = m_blk + wm + i * 16 + laneHi * 8 + r;
                C[(size_t)m * DIM + n] = acc[i][j][r];
            }
        }
}

extern "C" void kernel_launch(void* const* d_in, const int* in_sizes, int n_in,
                              void* d_out, int out_size, void* d_ws, size_t ws_size,
                              hipStream_t stream) {
    const float* x            = (const float*)d_in[0];   // [L, D]
    const float* x_mix_last   = (const float*)d_in[1];   // [D]
    const float* factor_param = (const float*)d_in[2];   // [D]
    const float* W            = (const float*)d_in[3];   // [D, D]
    float* out                = (float*)d_out;           // [L, D]

    char* ws = (char*)d_ws;
    uint16_t* xmix_bf16 = (uint16_t*)(ws);                               // 32 MB
    uint16_t* w_bf16    = (uint16_t*)(ws + 33554432);                    // 8 MB
    float*    carry     = (float*)   (ws + 33554432 + 8388608);          // 512 KB
    float*    in_state  = (float*)   (ws + 33554432 + 8388608 + 524288); // 512 KB

    ema_carry_kernel<<<dim3(DIM / 256, NCHUNK), 256, 0, stream>>>(x, factor_param, carry);
    ema_prefix_kernel<<<dim3(DIM / 256), 256, 0, stream>>>(x_mix_last, factor_param, carry, in_state);
    ema_apply_kernel<<<dim3(DIM / 256, NCHUNK), 256, 0, stream>>>(x, factor_param, in_state, xmix_bf16);
    bf16_convert4_kernel<<<dim3((DIM * DIM / 4) / 256), 256, 0, stream>>>(
        (const float4*)W, (ushort4*)w_bf16);

    const size_t lds_bytes = 4u * TILE_ELEMS * sizeof(uint16_t);   // 73728 B
    gemm_bf16_wmma_kernel<<<dim3(DIM / 128, L_SEQ / 128), 256, lds_bytes, stream>>>(
        xmix_bf16, w_bf16, out);
}
